// ProteinFeatures_81252191305793
// MI455X (gfx1250) — compile-verified, matrix operand order checked by reading, not run on hardware
//
#include <hip/hip_runtime.h>
#include <hip/hip_bf16.h>
#include <math.h>

// ---- problem constants (from reference setup) ----
#define LL     1536
#define TOPK   30
#define NATOM  14
#define NRBF   16
#define HID    128
#define NFEAT  3152          // 16 + 14*14*16
#define NKB64  50            // ceil(3152/64); K padded to 3200
#define ROWS   32            // edge rows per block (two 16-row WMMA tiles)

#define V_ELEMS   (LL * HID)                  // 196608
#define E_ELEMS   (LL * TOPK * HID)           // 5898240
#define I_ELEMS   (LL * TOPK)                 // 46080
#define X_ELEMS   (LL * NATOM * 3)            // 64512
#define EOFF      (V_ELEMS)
#define IOFF      (V_ELEMS + E_ELEMS)         // 6094848
#define XOFF      (IOFF + I_ELEMS)            // 6140928

// workspace layout
#define WS_EIDX_BYTES   (I_ELEMS * 4)         // 184320 (16B aligned)
#define WF16_ELEMS      (NKB64 * HID * 64)    // 409600 f16 (819200 B)

typedef __attribute__((ext_vector_type(16))) _Float16 v16h;
typedef __attribute__((ext_vector_type(8)))  _Float16 v8h;
typedef __attribute__((ext_vector_type(8)))  float    v8f;

// ============================================================
// Kernel 0: one-time W_edge f32 -> f16, tiled fragment-friendly:
// wf16[(kb*128 + n)*64 + kk] = W_edge[kb*64+kk][n], zero-padded K
// ============================================================
__global__ void __launch_bounds__(256)
wconv_kernel(const float* __restrict__ We, _Float16* __restrict__ wf16)
{
    int idx = blockIdx.x * 256 + threadIdx.x;
    if (idx >= WF16_ELEMS) return;
    int kb = idx >> 13;          // / (128*64)
    int r  = idx & 8191;
    int n  = r >> 6;
    int kk = r & 63;
    int f  = kb * 64 + kk;
    wf16[idx] = (_Float16)(f < NFEAT ? We[(size_t)f * HID + n] : 0.0f);
}

// ============================================================
// Kernel 1: masked pairwise Ca distances + top-30 selection
// ============================================================
__global__ void __launch_bounds__(256)
topk_kernel(const float* __restrict__ x, const float* __restrict__ mask,
            int* __restrict__ eidx, float* __restrict__ eidx_f)
{
    const int i   = blockIdx.x;
    const int tid = threadIdx.x;
    __shared__ float Dsh[LL];
    __shared__ float redv[256];
    __shared__ int   redi[256];

    const float cx = x[(i * NATOM + 1) * 3 + 0];
    const float cy = x[(i * NATOM + 1) * 3 + 1];
    const float cz = x[(i * NATOM + 1) * 3 + 2];
    const float mi = mask[i];

    for (int j = tid; j < LL; j += 256) {
        float dx = x[(j * NATOM + 1) * 3 + 0] - cx;
        float dy = x[(j * NATOM + 1) * 3 + 1] - cy;
        float dz = x[(j * NATOM + 1) * 3 + 2] - cz;
        float dist = sqrtf(dx * dx + dy * dy + dz * dz + 1e-6f);
        Dsh[j] = mi * mask[j] * dist;
    }
    __syncthreads();

    float mx = -1e30f;
    for (int j = tid; j < LL; j += 256) mx = fmaxf(mx, Dsh[j]);
    redv[tid] = mx;
    __syncthreads();
    for (int s = 128; s > 0; s >>= 1) {
        if (tid < s) redv[tid] = fmaxf(redv[tid], redv[tid + s]);
        __syncthreads();
    }
    const float Dmax = redv[0];
    __syncthreads();

    for (int j = tid; j < LL; j += 256) {
        float m2 = mi * mask[j];
        Dsh[j] += (1.0f - m2) * Dmax;
    }
    __syncthreads();

    for (int t = 0; t < TOPK; ++t) {
        float bv = 3.0e38f; int bj = 0x3fffffff;
        for (int j = tid; j < LL; j += 256) {
            float v = Dsh[j];
            if (v < bv || (v == bv && j < bj)) { bv = v; bj = j; }
        }
        redv[tid] = bv; redi[tid] = bj;
        __syncthreads();
        for (int s = 128; s > 0; s >>= 1) {
            if (tid < s) {
                float ov = redv[tid + s]; int oj = redi[tid + s];
                if (ov < redv[tid] || (ov == redv[tid] && oj < redi[tid])) {
                    redv[tid] = ov; redi[tid] = oj;
                }
            }
            __syncthreads();
        }
        if (tid == 0) {
            int sel = redi[0];
            eidx[i * TOPK + t]   = sel;
            eidx_f[i * TOPK + t] = (float)sel;
            Dsh[sel] = 3.0e38f;
        }
        __syncthreads();
    }
}

// ============================================================
// Kernel 2: node features  V = LN(concat(onehot(S), feat) @ W_node)
// ============================================================
__global__ void __launch_bounds__(128)
node_kernel(const float* __restrict__ feat, const float* __restrict__ Wn,
            const float* __restrict__ gn, const float* __restrict__ bn,
            const int* __restrict__ S, float* __restrict__ outV)
{
    const int i = blockIdx.x;
    const int c = threadIdx.x;
    __shared__ float sf[HID];
    __shared__ float sh[HID];

    sf[c] = feat[i * HID + c];
    __syncthreads();

    float acc = Wn[S[i] * HID + c];
    for (int f = 0; f < HID; ++f)
        acc += sf[f] * Wn[(21 + f) * HID + c];

    sh[c] = acc;
    __syncthreads();
    for (int s = 64; s > 0; s >>= 1) { if (c < s) sh[c] += sh[c + s]; __syncthreads(); }
    const float sum = sh[0];
    __syncthreads();
    sh[c] = acc * acc;
    __syncthreads();
    for (int s = 64; s > 0; s >>= 1) { if (c < s) sh[c] += sh[c + s]; __syncthreads(); }
    const float sumsq = sh[0];

    const float mean = sum * (1.0f / 128.0f);
    const float var  = sumsq * (1.0f / 128.0f) - mean * mean;
    const float rstd = rsqrtf(var + 1e-5f);
    outV[i * HID + c] = (acc - mean) * rstd * gn[c] + bn[c];
}

// ============================================================
// Kernel 3: edge features — fused RBF + positional + WMMA GEMM + LN
// block = 256 threads (8 waves) -> 32 edge rows x 128 cols
//   * B tiles via async DMA (global->LDS), double-buffered
//   * branch-free A generation through a {pair,rbf} LUT, mask folded
//     into Dp as +inf (exp -> 0), K-padding via sentinel column 196
//   * LN tile aliases dead B buffer 0 (last iter reads buffer 1)
// ============================================================
struct EdgeSmem {
    int       sI[ROWS], sJ[ROWS];
    _Float16  sDp[ROWS][197];              // col 196 = +inf sentinel
    float     sEpos[ROWS][16];
    __align__(16) _Float16 sA[ROWS][64];   // A tile 32x64 f16
    unsigned short lut[3200];              // f -> p | (r<<8)
    union {
        __align__(16) _Float16 sBt[2][HID][64];  // double-buffered B^T (32 KB)
        float sTile[ROWS][HID];                  // 16 KB, aliases buffer 0
    } u;
};

__global__ void __launch_bounds__(256)
edge_kernel(const float* __restrict__ x, const float* __restrict__ mask,
            const float* __restrict__ am, const int* __restrict__ Ridx,
            const int* __restrict__ chain,
            const float* __restrict__ Wpos, const float* __restrict__ bpos,
            const _Float16* __restrict__ wf16,
            const float* __restrict__ ge, const float* __restrict__ be,
            const int* __restrict__ eidx, float* __restrict__ outE)
{
    const int tid = threadIdx.x;
    const int r0  = blockIdx.x * ROWS;
    __shared__ EdgeSmem sm;

    if (tid < ROWS) {
        int e = r0 + tid;
        sm.sI[tid] = e / TOPK;
        sm.sJ[tid] = eidx[e];
    }
    // feature LUT: f -> atom-pair index p, rbf index r
    for (int f = tid; f < 3200; f += 256) {
        unsigned p = 196, r = 0;
        if (f >= 16 && f < NFEAT) {
            int g = f - 16;
            int a = g / 224;                 // 224 = 14*16
            int rem = g - 224 * a;
            p = a * NATOM + (rem >> 4);
            r = rem & 15;
        }
        sm.lut[f] = (unsigned short)(p | (r << 8));
    }
    __syncthreads();   // sI/sJ ready before Dp pass

    for (int idx = tid; idx < ROWS * 197; idx += 256) {
        int m = idx / 197, p = idx - 197 * m;
        float d = __builtin_inff();
        if (p < 196) {
            int a = p / NATOM, b = p - NATOM * a;
            int i = sm.sI[m], j = sm.sJ[m];
            const float* pi = x + (size_t)(i * NATOM + a) * 3;
            const float* pj = x + (size_t)(j * NATOM + b) * 3;
            float dx = pi[0] - pj[0], dy = pi[1] - pj[1], dz = pi[2] - pj[2];
            float dist = sqrtf(dx * dx + dy * dy + dz * dz + 1e-6f);
            float msk  = am[i * NATOM + a] * mask[i] *
                         am[j * NATOM + b] * mask[j];
            d = (msk != 0.0f) ? dist : __builtin_inff();
        }
        sm.sDp[m][p] = (_Float16)d;
    }
    for (int idx = tid; idx < ROWS * 16; idx += 256) {
        int m = idx >> 4, f = idx & 15;
        int i = sm.sI[m], j = sm.sJ[m];
        int off = Ridx[i] - Ridx[j];
        int same = (chain[i] == chain[j]);
        int dpos = same ? min(max(off + 32, 0), 64) : 65;
        sm.sEpos[m][f] = Wpos[dpos * 16 + f] + bpos[f];
    }

    const int wv = tid >> 5;      // wave 0..7 -> output cols [16*wv,16*wv+16)
    const int ln = tid & 31;
    const int hi = ln >> 4;       // K-half selector (WMMA 16-bit layout)
    const int lr = ln & 15;       // A row / B col within tile

    // async DMA one 128x64 f16 tile (16 KB): 64 B per thread = 4x b128
    auto issue_tile = [&](int kb64, int buf) {
        unsigned lds = (unsigned)(unsigned long long)(&sm.u.sBt[buf][0][0]) +
                       (unsigned)tid * 64u;
        unsigned long long g =
            (unsigned long long)(wf16 + (size_t)kb64 * (HID * 64)) +
            (unsigned long long)tid * 64ull;
        for (int c = 0; c < 4; ++c) {
            asm volatile("global_load_async_to_lds_b128 %0, %1, off"
                         :: "v"(lds + (unsigned)(c * 16)),
                            "v"(g + (unsigned long long)(c * 16))
                         : "memory");
        }
    };

    issue_tile(0, 0);
    __syncthreads();              // Dp/Epos/lut ready

    v8f acc0 = {}, acc1 = {};
    for (int kb = 0; kb < NKB64; ++kb) {
        const int cb = kb & 1, nb = cb ^ 1;
        if (kb + 1 < NKB64) issue_tile(kb + 1, nb);   // prefetch next B tile

        // generate A tile: 8 contiguous f16 per thread
        const int fbase = kb * 64;
        const int m  = tid >> 3;
        const int k0 = (tid * 8) & 63;
        if (kb == 0) {            // only tile containing positional features
            for (int q = 0; q < 8; ++q) {
                int kk = k0 + q;
                float v;
                if (kk < 16) {
                    v = sm.sEpos[m][kk];
                } else {
                    unsigned pr = sm.lut[kk];
                    float d  = (float)sm.sDp[m][pr & 255];
                    float mu = (float)(pr >> 8) * (20.0f / 15.0f);
                    float t  = (d - mu) * 0.8f;
                    v = __expf(-t * t);
                }
                sm.sA[m][kk] = (_Float16)v;
            }
        } else {                  // steady state: branch-free
            for (int q = 0; q < 8; ++q) {
                int kk = k0 + q;
                unsigned pr = sm.lut[fbase + kk];
                float d  = (float)sm.sDp[m][pr & 255];
                float mu = (float)(pr >> 8) * (20.0f / 15.0f);
                float t  = (d - mu) * 0.8f;
                sm.sA[m][kk] = (_Float16)__expf(-t * t);
            }
        }

        if (kb + 1 < NKB64) asm volatile("s_wait_asynccnt 4" ::: "memory");
        else                asm volatile("s_wait_asynccnt 0" ::: "memory");
        __syncthreads();

        // 4 WMMAs: 2 K-subblocks x 2 row tiles (B fragment shared)
        #pragma unroll
        for (int t = 0; t < 2; ++t) {
            v16h bvec, avec0, avec1;
            {
                const v8h b0 = *(const v8h*)&sm.u.sBt[cb][wv * 16 + lr][t * 32 + hi * 8];
                const v8h b1 = *(const v8h*)&sm.u.sBt[cb][wv * 16 + lr][t * 32 + 16 + hi * 8];
                const v8h a0 = *(const v8h*)&sm.sA[lr][t * 32 + hi * 8];
                const v8h a1 = *(const v8h*)&sm.sA[lr][t * 32 + 16 + hi * 8];
                const v8h c0 = *(const v8h*)&sm.sA[16 + lr][t * 32 + hi * 8];
                const v8h c1 = *(const v8h*)&sm.sA[16 + lr][t * 32 + 16 + hi * 8];
                for (int e2 = 0; e2 < 8; ++e2) {
                    bvec[e2] = b0[e2]; bvec[e2 + 8] = b1[e2];
                    avec0[e2] = a0[e2]; avec0[e2 + 8] = a1[e2];
                    avec1[e2] = c0[e2]; avec1[e2 + 8] = c1[e2];
                }
            }
            acc0 = __builtin_amdgcn_wmma_f32_16x16x32_f16(
                false, avec0, false, bvec, (short)0, acc0, false, false);
            acc1 = __builtin_amdgcn_wmma_f32_16x16x32_f16(
                false, avec1, false, bvec, (short)0, acc1, false, false);
        }
        __syncthreads();
    }

    // C/D layout: vgpr v -> M = v + 8*hi, N = lr (+16*wv); rows 16..31 from acc1
    for (int v2 = 0; v2 < 8; ++v2) {
        sm.u.sTile[v2 + 8 * hi][wv * 16 + lr]      = acc0[v2];
        sm.u.sTile[16 + v2 + 8 * hi][wv * 16 + lr] = acc1[v2];
    }
    __syncthreads();

    // per-row layernorm: 8 threads/row, 16 values each, width-8 shuffles
    {
        int row = tid >> 3, cg = tid & 7;
        float vals[16];
        float s = 0.0f, sq = 0.0f;
        for (int q = 0; q < 16; ++q) {
            float vv = sm.u.sTile[row][cg * 16 + q];
            vals[q] = vv; s += vv; sq += vv * vv;
        }
        for (int o = 4; o > 0; o >>= 1) {
            s  += __shfl_xor(s,  o, 8);
            sq += __shfl_xor(sq, o, 8);
        }
        const float mean = s * (1.0f / 128.0f);
        const float var  = sq * (1.0f / 128.0f) - mean * mean;
        const float rstd = rsqrtf(var + 1e-5f);
        const size_t grow = (size_t)(r0 + row);
        for (int q = 0; q < 16; ++q) {
            int c = cg * 16 + q;
            outE[grow * HID + c] = (vals[q] - mean) * rstd * ge[c] + be[c];
        }
    }
}

// ============================================================
extern "C" void kernel_launch(void* const* d_in, const int* in_sizes, int n_in,
                              void* d_out, int out_size, void* d_ws, size_t ws_size,
                              hipStream_t stream)
{
    (void)in_sizes; (void)n_in; (void)out_size; (void)ws_size;

    const float* x     = (const float*)d_in[0];
    const float* mask  = (const float*)d_in[1];
    const float* am    = (const float*)d_in[2];
    const float* feat  = (const float*)d_in[3];
    const float* Wpos  = (const float*)d_in[4];
    const float* bpos  = (const float*)d_in[5];
    const float* Wedge = (const float*)d_in[6];
    const float* ge    = (const float*)d_in[7];
    const float* be    = (const float*)d_in[8];
    const float* Wn    = (const float*)d_in[9];
    const float* gn    = (const float*)d_in[10];
    const float* bn    = (const float*)d_in[11];
    const int*   S     = (const int*)d_in[12];
    const int*   Ridx  = (const int*)d_in[13];
    const int*   chain = (const int*)d_in[14];

    float*    out  = (float*)d_out;
    int*      eidx = (int*)d_ws;
    _Float16* wf16 = (_Float16*)((char*)d_ws + WS_EIDX_BYTES);

    float* outV = out;
    float* outE = out + EOFF;
    float* outI = out + IOFF;
    float* outX = out + XOFF;

    wconv_kernel<<<(WF16_ELEMS + 255) / 256, 256, 0, stream>>>(Wedge, wf16);
    topk_kernel<<<LL, 256, 0, stream>>>(x, mask, eidx, outI);
    node_kernel<<<LL, 128, 0, stream>>>(feat, Wn, gn, bn, S, outV);
    edge_kernel<<<(LL * TOPK) / ROWS, 256, 0, stream>>>(
        x, mask, am, Ridx, chain, Wpos, bpos, wf16, ge, be, eidx, outE);
    hipMemcpyAsync(outX, x, (size_t)X_ELEMS * sizeof(float),
                   hipMemcpyDeviceToDevice, stream);
}
